// linerAttention_72748156059772
// MI455X (gfx1250) — matrix operand order repaired and checked by reading, not compiled, not run on hardware
//
#include <hip/hip_runtime.h>
#include <hip/hip_bf16.h>

// ---------------------------------------------------------------------------
// XCiT-style linear attention block for MI455X (gfx1250, wave32, WMMA bf16).
// cvt -> GEMM(qkv, TDM-staged) -> dwconv(split heads) -> attn -> GEMM(out).
// ---------------------------------------------------------------------------

typedef __bf16 v16bf __attribute__((ext_vector_type(16)));
typedef __bf16 v8bf  __attribute__((ext_vector_type(8)));
typedef float  v8f   __attribute__((ext_vector_type(8)));
typedef unsigned int u32x4 __attribute__((ext_vector_type(4)));
typedef int i32x4 __attribute__((ext_vector_type(4)));
typedef int i32x8 __attribute__((ext_vector_type(8)));

#define WMMA_BF16(a, b, c) \
  __builtin_amdgcn_wmma_f32_16x16x32_bf16(false, (a), false, (b), (short)0, (c), false, false)

// Tensor Data Mover: builtin exists on both toolchains but with different
// arity. The therock headers (amd_gfx1250_TDM.h) ship only with the 6-arg
// toolchain, so use header presence as the arity detector.
#if defined(__gfx1250__) && __has_builtin(__builtin_amdgcn_tensor_load_to_lds)
#  define USE_TDM 1
#  if __has_include(<hip/amd_detail/amd_gfx1250_TDM.h>)
#    define TDM_ARITY 6
#  else
#    define TDM_ARITY 5
#  endif
#else
#  define USE_TDM 0
#endif

static __device__ __forceinline__ v16bf cat16(v8bf lo, v8bf hi) {
  return __builtin_shufflevector(lo, hi, 0,1,2,3,4,5,6,7,8,9,10,11,12,13,14,15);
}

// A-matrix fragment (16x32 bf16, ISA 7.12.2): lane = row M, element e ->
// K = e + 8*half (e<8) or 16 + (e-8) + 8*half. rp = row start of K chunk.
static __device__ __forceinline__ v16bf frag_A(const __bf16* rp, int half) {
  v8bf lo = *(const v8bf*)(rp + 8 * half);
  v8bf hi = *(const v8bf*)(rp + 16 + 8 * half);
  return cat16(lo, hi);
}

// B-matrix fragment (32x16 bf16): lane = column N, element e ->
// K = e + 8*half (e<8) or 16 + (e-8) + 8*half.
// rp points at row N of a [N][K] transposed tile.
static __device__ __forceinline__ v16bf frag_B(const __bf16* rp, int half) {
  v8bf lo = *(const v8bf*)(rp + 8 * half);
  v8bf hi = *(const v8bf*)(rp + 16 + 8 * half);
  return cat16(lo, hi);
}

#if USE_TDM
// Issue a TDM 2-D tile load (tile_d0 x tile_d1 elements of 2 bytes, row
// stride in elements) from global into LDS (compact row-major). D# packing
// per ISA 8.3-8.4: group0 = {count, lds_addr, global_addr, type=2};
// group1 = {data_size=2B, tensor dims, tile dims, dim0 stride}.
static __device__ __forceinline__ void tdm_load_2d_bf16(unsigned lds_byte_addr,
                                                        const void* gaddr,
                                                        unsigned tile_d0, unsigned tile_d1,
                                                        unsigned stride_elems) {
  unsigned long long ga = (unsigned long long)(size_t)gaddr;
  u32x4 g0;
  g0[0] = 1u;                                             // count=1, user D#
  g0[1] = lds_byte_addr;                                  // lds_addr[31:0]
  g0[2] = (unsigned)(ga & 0xffffffffull);                 // global_addr[31:0]
  g0[3] = (unsigned)((ga >> 32) & 0x1ffffffull) | (2u << 30);  // [56:32] | type=2
  const unsigned td0 = 0x100000u, td1 = 0x100000u;        // large: no OOB clamp
  i32x8 g1;
  g1[0] = (int)(1u << 16);                                // data_size=1 (2B)
  g1[1] = (int)((td0 & 0xffffu) << 16);                   // tensor_dim0 lo16
  g1[2] = (int)((td0 >> 16) | ((td1 & 0xffffu) << 16));   // td0 hi / td1 lo
  g1[3] = (int)((td1 >> 16) | (tile_d0 << 16));           // td1 hi / tile_dim0
  g1[4] = (int)(tile_d1 & 0xffffu);                       // tile_dim1, tile_dim2=0
  g1[5] = (int)stride_elems;                              // tensor_dim0_stride lo32
  g1[6] = 0;
  g1[7] = 0;
  i32x4 z4 = {0, 0, 0, 0};                                // 2-D: groups 2/3 unused
#if TDM_ARITY == 6
  i32x8 z8 = {0, 0, 0, 0, 0, 0, 0, 0};
  __builtin_amdgcn_tensor_load_to_lds(g0, g1, z4, z4, z8, 0);
#else
  __builtin_amdgcn_tensor_load_to_lds(g0, g1, z4, z4, 0);
#endif
}
static __device__ __forceinline__ unsigned lds_off(const void* p) {
  return (unsigned)(unsigned long long)(size_t)p;         // LDS aperture: off in [31:0]
}
#endif

// ---------------------------------------------------------------------------
// Converters
// ---------------------------------------------------------------------------
__global__ void cvt_f32_bf16_v8(const float* __restrict__ in, __bf16* __restrict__ out,
                                long n8) {
  long i = (long)blockIdx.x * blockDim.x + threadIdx.x;
  if (i >= n8) return;
  const float4* p = (const float4*)in + i * 2;
  float4 a = p[0], b = p[1];
  v8bf o;
  o[0] = (__bf16)a.x; o[1] = (__bf16)a.y; o[2] = (__bf16)a.z; o[3] = (__bf16)a.w;
  o[4] = (__bf16)b.x; o[5] = (__bf16)b.y; o[6] = (__bf16)b.z; o[7] = (__bf16)b.w;
  *(v8bf*)(out + i * 8) = o;
}

// out[n*K + k] = in[k*N + n]  (weight transposed, bf16)
__global__ void cvt_transpose_bf16(const float* __restrict__ in, __bf16* __restrict__ out,
                                   int K, int N) {
  long i = (long)blockIdx.x * blockDim.x + threadIdx.x;
  if (i < (long)K * N) {
    int n = (int)(i / K), kk = (int)(i % K);
    out[i] = (__bf16)in[(long)kk * N + n];
  }
}

// ---------------------------------------------------------------------------
// WMMA tile-compute shared by both GEMM staging paths.
// ---------------------------------------------------------------------------
static __device__ __forceinline__ void mma_128x128(const __bf16* As, const __bf16* Bs,
                                                   v8f (&acc)[2][4], int wm, int wn,
                                                   int lane, int half) {
  v16bf bfr[4];
#pragma unroll
  for (int ni = 0; ni < 4; ++ni)
    bfr[ni] = frag_B(&Bs[(wn * 64 + ni * 16 + (lane & 15)) * 32], half);
#pragma unroll
  for (int mi = 0; mi < 2; ++mi) {
    v16bf af = frag_A(&As[(wm * 32 + mi * 16 + (lane & 15)) * 32], half);
#pragma unroll
    for (int ni = 0; ni < 4; ++ni)
      acc[mi][ni] = WMMA_BF16(af, bfr[ni], acc[mi][ni]);
  }
}

// ---------------------------------------------------------------------------
// Tiled WMMA GEMM: C[M,NDIM] = A[M,512] * B[512,NDIM] + bias, B as BT[NDIM,512].
// 256 threads (8 waves), 128x128 tile, K chunks of 32 through double-buffered
// LDS. TDM path: wave 0 DMAs the next A/B tiles (tensor_load_to_lds) while all
// waves compute; s_wait_tensorcnt + barrier for hand-off.
// ---------------------------------------------------------------------------
template <int NDIM, bool OUT_BF16>
__launch_bounds__(256)
__global__ void gemm_k512_bf16(const __bf16* __restrict__ A,
                               const __bf16* __restrict__ BT,
                               const float* __restrict__ bias,
                               void* __restrict__ out) {
  __shared__ __align__(16) __bf16 As[2][128 * 32];
  __shared__ __align__(16) __bf16 Bs[2][128 * 32];

  const int tid  = threadIdx.x;
  const int lane = tid & 31;
  const int half = lane >> 4;
  const int wid  = tid >> 5;
  const int wm   = wid & 3;
  const int wn   = wid >> 2;
  const int m0   = blockIdx.y * 128;
  const int n0   = blockIdx.x * 128;

  v8f acc[2][4];
#pragma unroll
  for (int mi = 0; mi < 2; ++mi)
#pragma unroll
    for (int ni = 0; ni < 4; ++ni) acc[mi][ni] = (v8f)0.0f;

#if USE_TDM
  const bool issuer = (wid == 0);
  if (issuer) {
    tdm_load_2d_bf16(lds_off(&As[0][0]), A  + (long)m0 * 512, 32, 128, 512);
    tdm_load_2d_bf16(lds_off(&Bs[0][0]), BT + (long)n0 * 512, 32, 128, 512);
  }
#pragma unroll 1
  for (int kt = 0; kt < 16; ++kt) {
    if (issuer) {
      if (kt < 15) {
        const int k1 = (kt + 1) * 32, nb = (kt + 1) & 1;
        tdm_load_2d_bf16(lds_off(&As[nb][0]), A  + (long)m0 * 512 + k1, 32, 128, 512);
        tdm_load_2d_bf16(lds_off(&Bs[nb][0]), BT + (long)n0 * 512 + k1, 32, 128, 512);
        __builtin_amdgcn_s_wait_tensorcnt((short)2);   // pair(kt) landed in LDS
      } else {
        __builtin_amdgcn_s_wait_tensorcnt((short)0);
      }
    }
    __syncthreads();
    mma_128x128(As[kt & 1], Bs[kt & 1], acc, wm, wn, lane, half);
    __syncthreads();
  }
#else
  const int srow = tid >> 1;
  const int sseg = (tid & 1) * 16;
#pragma unroll 1
  for (int kt = 0; kt < 16; ++kt) {
    const int k0 = kt * 32;
    const __bf16* ga = A  + (long)(m0 + srow) * 512 + k0 + sseg;
    const __bf16* gb = BT + (long)(n0 + srow) * 512 + k0 + sseg;
    *(v8bf*)&As[0][srow * 32 + sseg]     = *(const v8bf*)ga;
    *(v8bf*)&As[0][srow * 32 + sseg + 8] = *(const v8bf*)(ga + 8);
    *(v8bf*)&Bs[0][srow * 32 + sseg]     = *(const v8bf*)gb;
    *(v8bf*)&Bs[0][srow * 32 + sseg + 8] = *(const v8bf*)(gb + 8);
    if (kt < 15) {
      __builtin_prefetch(ga + 32, 0, 1);
      __builtin_prefetch(gb + 32, 0, 1);
    }
    __syncthreads();
    mma_128x128(As[0], Bs[0], acc, wm, wn, lane, half);
    __syncthreads();
  }
#endif

  // Epilogue: C/D layout -> lane col = lane&15, rows r + 8*half.
#pragma unroll
  for (int mi = 0; mi < 2; ++mi)
#pragma unroll
    for (int ni = 0; ni < 4; ++ni) {
      const int col = n0 + wn * 64 + ni * 16 + (lane & 15);
      const float bv = bias[col];
      const int rbase = m0 + wm * 32 + mi * 16 + 8 * half;
#pragma unroll
      for (int r = 0; r < 8; ++r) {
        const float val = acc[mi][ni][r] + bv;
        const long idx = (long)(rbase + r) * NDIM + col;
        if constexpr (OUT_BF16) ((__bf16*)out)[idx] = (__bf16)val;
        else                    ((float*)out)[idx]  = val;
      }
    }
}

// ---------------------------------------------------------------------------
// Depthwise 3x3 conv (SAME) + bias, splitting q/k/v into head-major
// [B,H,N,D] bf16. 8 channels per thread -> b128 loads/stores.
// qkv: [B,64,64,1536] bf16. wdw: [3,3,1,1536] f32.
// ---------------------------------------------------------------------------
__global__ void dwconv_split_v8(const __bf16* __restrict__ qkv, const float* __restrict__ wdw,
                                const float* __restrict__ bdw, __bf16* __restrict__ qkvh) {
  const long idx = (long)blockIdx.x * blockDim.x + threadIdx.x;
  if (idx >= (long)16 * 4096 * 192) return;
  const int c  = (int)(idx % 192) * 8;       // 8 consecutive channels
  const long pix = idx / 192;
  const int xx = (int)(pix & 63);
  const int yy = (int)((pix >> 6) & 63);
  const int b  = (int)(pix >> 12);

  float acc[8];
  {
    float4 b0 = *(const float4*)(bdw + c), b1 = *(const float4*)(bdw + c + 4);
    acc[0] = b0.x; acc[1] = b0.y; acc[2] = b0.z; acc[3] = b0.w;
    acc[4] = b1.x; acc[5] = b1.y; acc[6] = b1.z; acc[7] = b1.w;
  }
#pragma unroll
  for (int dy = -1; dy <= 1; ++dy) {
    const int y2 = yy + dy;
    if ((unsigned)y2 >= 64u) continue;
#pragma unroll
    for (int dx = -1; dx <= 1; ++dx) {
      const int x2 = xx + dx;
      if ((unsigned)x2 >= 64u) continue;
      const v8bf iv = *(const v8bf*)&qkv[(((long)b * 64 + y2) * 64 + x2) * 1536 + c];
      const float* wp = wdw + ((dy + 1) * 3 + (dx + 1)) * 1536 + c;
      float4 w0 = *(const float4*)wp, w1 = *(const float4*)(wp + 4);
      acc[0] += (float)iv[0] * w0.x; acc[1] += (float)iv[1] * w0.y;
      acc[2] += (float)iv[2] * w0.z; acc[3] += (float)iv[3] * w0.w;
      acc[4] += (float)iv[4] * w1.x; acc[5] += (float)iv[5] * w1.y;
      acc[6] += (float)iv[6] * w1.z; acc[7] += (float)iv[7] * w1.w;
    }
  }
  const int g  = c / 512;                    // 0=q 1=k 2=v
  const int cc = c % 512;
  const int h  = cc >> 6, d = cc & 63;       // d multiple of 8: aligned b128 store
  const int n  = (int)(pix & 4095);
  v8bf o;
#pragma unroll
  for (int i = 0; i < 8; ++i) o[i] = (__bf16)acc[i];
  *(v8bf*)&qkvh[(long)g * 33554432 + ((long)(b * 8 + h) * 4096 + n) * 64 + d] = o;
}

// ---------------------------------------------------------------------------
// XCiT attention per (b,h): dots = scale * k^T q (64x64, K=4096) via WMMA with
// LDS-transposed 32-n chunks; softmax in LDS writing attn^T bf16; then
// out = v @ attn with A fragments straight from global. 128 threads = 4 waves.
// ---------------------------------------------------------------------------
__launch_bounds__(128)
__global__ void xcit_attn(const __bf16* __restrict__ q, const __bf16* __restrict__ k,
                          const __bf16* __restrict__ v, __bf16* __restrict__ ao) {
  __shared__ __align__(16) __bf16 Kt[64 * 32];
  __shared__ __align__(16) __bf16 Qt[64 * 32];
  __shared__ __align__(16) float  Ds[64 * 64];
  __shared__ __align__(16) __bf16 At[64 * 64];

  const int tid = threadIdx.x, lane = tid & 31, wid = tid >> 5, half = lane >> 4;
  const int bh = blockIdx.x;
  const int b = bh >> 3, h = bh & 7;
  const __bf16* kh = k + (long)bh * 4096 * 64;
  const __bf16* qh = q + (long)bh * 4096 * 64;
  const __bf16* vh = v + (long)bh * 4096 * 64;

  v8f acc[4];
#pragma unroll
  for (int i = 0; i < 4; ++i) acc[i] = (v8f)0.0f;

  const int nl = tid & 31;
  const int d0 = (tid >> 5) * 16;

#pragma unroll 1
  for (int n0 = 0; n0 < 4096; n0 += 32) {
    const __bf16* gk = kh + (long)(n0 + nl) * 64 + d0;
    const __bf16* gq = qh + (long)(n0 + nl) * 64 + d0;
    v8bf kv0 = *(const v8bf*)gk, kv1 = *(const v8bf*)(gk + 8);
    v8bf qv0 = *(const v8bf*)gq, qv1 = *(const v8bf*)(gq + 8);
#pragma unroll
    for (int ii = 0; ii < 8; ++ii) {
      Kt[(d0 + ii) * 32 + nl]     = kv0[ii];
      Kt[(d0 + 8 + ii) * 32 + nl] = kv1[ii];
      Qt[(d0 + ii) * 32 + nl]     = qv0[ii];
      Qt[(d0 + 8 + ii) * 32 + nl] = qv1[ii];
    }
    __syncthreads();
    v16bf af = frag_A(&Kt[(wid * 16 + (lane & 15)) * 32], half);
#pragma unroll
    for (int nj = 0; nj < 4; ++nj) {
      v16bf bfr = frag_B(&Qt[(nj * 16 + (lane & 15)) * 32], half);
      acc[nj] = WMMA_BF16(af, bfr, acc[nj]);
    }
    __syncthreads();
  }

#pragma unroll
  for (int nj = 0; nj < 4; ++nj)
#pragma unroll
    for (int r = 0; r < 8; ++r)
      Ds[(wid * 16 + r + 8 * half) * 64 + nj * 16 + (lane & 15)] = acc[nj][r] * 0.125f;
  __syncthreads();

  if (tid < 64) {
    float mx = Ds[tid * 64];
    for (int j = 1; j < 64; ++j) mx = fmaxf(mx, Ds[tid * 64 + j]);
    float s = 0.f;
    for (int j = 0; j < 64; ++j) s += __expf(Ds[tid * 64 + j] - mx);
    const float inv = 1.0f / s;
    for (int j = 0; j < 64; ++j)
      At[j * 64 + tid] = (__bf16)(__expf(Ds[tid * 64 + j] - mx) * inv);
  }
  __syncthreads();

#pragma unroll 1
  for (int it = 0; it < 64; ++it) {
    const int nb = (it * 4 + wid) * 16;
    v8f o[4];
#pragma unroll
    for (int et = 0; et < 4; ++et) o[et] = (v8f)0.0f;
#pragma unroll
    for (int ks = 0; ks < 2; ++ks) {
      v16bf af = frag_A(vh + (long)(nb + (lane & 15)) * 64 + ks * 32, half);
#pragma unroll
      for (int et = 0; et < 4; ++et) {
        v16bf bfr = frag_B(&At[(et * 16 + (lane & 15)) * 64 + ks * 32], half);
        o[et] = WMMA_BF16(af, bfr, o[et]);
      }
    }
#pragma unroll
    for (int et = 0; et < 4; ++et) {
      const int cc = h * 64 + et * 16 + (lane & 15);
#pragma unroll
      for (int r = 0; r < 8; ++r) {
        const long rowg = (long)b * 4096 + nb + r + 8 * half;
        ao[rowg * 512 + cc] = (__bf16)o[et][r];
      }
    }
  }
}

// ---------------------------------------------------------------------------
// Host launcher
// ---------------------------------------------------------------------------
extern "C" void kernel_launch(void* const* d_in, const int* in_sizes, int n_in,
                              void* d_out, int out_size, void* d_ws, size_t ws_size,
                              hipStream_t stream) {
  (void)in_sizes; (void)n_in; (void)out_size; (void)ws_size;
  const float* x     = (const float*)d_in[0];   // [16,4096,512]
  const float* w_qkv = (const float*)d_in[1];   // [512,1536]
  const float* b_qkv = (const float*)d_in[2];   // [1536]
  const float* w_dw  = (const float*)d_in[3];   // [3,3,1,1536]
  const float* b_dw  = (const float*)d_in[4];   // [1536]
  const float* w_out = (const float*)d_in[5];   // [512,512]
  const float* b_out = (const float*)d_in[6];   // [512]
  float* out = (float*)d_out;                   // [16,4096,512] f32

  char* ws = (char*)d_ws;
  __bf16* xb    = (__bf16*)(ws);                      //  67,108,864 B
  __bf16* wqkvT = (__bf16*)(ws + 67108864);           //   1,572,864 B
  __bf16* woutT = (__bf16*)(ws + 68681728);           //     524,288 B
  __bf16* qkvb  = (__bf16*)(ws + 69206016);           // 201,326,592 B  [B,64,64,1536]
  __bf16* qkvh  = (__bf16*)(ws + 270532608);          // 201,326,592 B  3x[B*H,4096,64]
  __bf16* ao    = (__bf16*)(ws + 471859200);          //  67,108,864 B  [B*N,512]

  cvt_f32_bf16_v8<<<16384, 256, 0, stream>>>(x, xb, 4194304L);
  cvt_transpose_bf16<<<3072, 256, 0, stream>>>(w_qkv, wqkvT, 512, 1536);
  cvt_transpose_bf16<<<1024, 256, 0, stream>>>(w_out, woutT, 512, 512);

  gemm_k512_bf16<1536, true><<<dim3(12, 512), 256, 0, stream>>>(xb, wqkvT, b_qkv, qkvb);

  dwconv_split_v8<<<49152, 256, 0, stream>>>(qkvb, w_dw, b_dw, qkvh);

  xcit_attn<<<128, 128, 0, stream>>>(qkvh, qkvh + 33554432L, qkvh + 67108864L, ao);

  gemm_k512_bf16<512, false><<<dim3(4, 512), 256, 0, stream>>>(ao, woutT, b_out, out);
}
